// SkipConnection_5085241278491
// MI455X (gfx1250) — matrix operand order repaired and checked by reading, not compiled, not run on hardware
//
#include <hip/hip_runtime.h>

#define B_ 4
#define N_ 1024
#define C_ 768
#define H_ 8
#define D_ 96
#define BN_EPS 1e-5f

typedef __attribute__((ext_vector_type(16))) __bf16 v16bf;
typedef __attribute__((ext_vector_type(8)))  __bf16 v8bf;
typedef __attribute__((ext_vector_type(8)))  float  v8f;

__device__ inline unsigned short f2bf(float f) {
  unsigned int u = __builtin_bit_cast(unsigned int, f);
  u += 0x7fffu + ((u >> 16) & 1u);            // round-to-nearest-even
  return (unsigned short)(u >> 16);
}
__device__ inline float bf2f(unsigned short s) {
  unsigned int u = ((unsigned int)s) << 16;
  return __builtin_bit_cast(float, u);
}

// A-fragment (16x32 bf16, M x K): lane row = lane%16; half = lane/16.
// e[i] = K = c0 + 16*half + i
__device__ inline v16bf ldA(const unsigned short* row, int c0, int half) {
  return *reinterpret_cast<const v16bf*>(row + c0 + 16 * half);
}
// B-fragment (32x16 bf16, K x N): lane col = lane%16; e[i] = K = c0 + 16*half + i
// -> one contiguous 32B load when B is stored K-contiguous per column.
__device__ inline v16bf ldB(const unsigned short* row, int c0, int half) {
  return *reinterpret_cast<const v16bf*>(row + c0 + 16 * half);
}
__device__ inline v8f wmma_bf16(v16bf a, v16bf b, v8f c) {
  return __builtin_amdgcn_wmma_f32_16x16x32_bf16(false, a, false, b, (short)0, c,
                                                 false, false);
}

// ---------------- K1: per-token 3x3 conv (3 ch, 16x16 img) + bf16 cast ----------
__global__ __launch_bounds__(256) void conv_cast_kernel(
    const float* __restrict__ q, const float* __restrict__ k,
    const float* __restrict__ v, const float* __restrict__ qw,
    const float* __restrict__ kw, const float* __restrict__ vw,
    unsigned short* __restrict__ qb, unsigned short* __restrict__ kb,
    unsigned short* __restrict__ vt) {
  __shared__ float sh[3 * 768];
  __shared__ float wsh[3 * 81];
  int t = threadIdx.x;
  int tok = blockIdx.x;  // b*N + n
  int b = tok / N_, n = tok % N_;
  const float* src[3]  = {q + (size_t)tok * C_, k + (size_t)tok * C_, v + (size_t)tok * C_};
  const float* wsrc[3] = {qw, kw, vw};
  for (int i = t; i < 3 * 768; i += 256) sh[i] = src[i / 768][i % 768];
  for (int i = t; i < 3 * 81; i += 256) wsh[i] = wsrc[i / 81][i % 81];
  __syncthreads();
  int py = t >> 4, px = t & 15;
  for (int ts = 0; ts < 3; ++ts) {
    for (int oc = 0; oc < 3; ++oc) {
      float acc = 0.f;
      for (int ic = 0; ic < 3; ++ic)
        for (int ky = 0; ky < 3; ++ky) {
          int yy = py + ky - 1;
          if (yy < 0 || yy > 15) continue;
          for (int kx = 0; kx < 3; ++kx) {
            int xx = px + kx - 1;
            if (xx < 0 || xx > 15) continue;
            acc += sh[ts * 768 + ic * 256 + yy * 16 + xx] *
                   wsh[ts * 81 + ((oc * 3 + ic) * 3 + ky) * 3 + kx];
          }
        }
      int c = oc * 256 + t;       // flat conv-output channel
      int h = c / D_, dch = c % D_;
      unsigned short val = f2bf(acc);
      if (ts == 0)      qb[((size_t)(b * H_ + h) * N_ + n) * D_ + dch] = val;
      else if (ts == 1) kb[((size_t)(b * H_ + h) * N_ + n) * D_ + dch] = val;
      else              vt[((size_t)(b * H_ + h) * D_ + dch) * N_ + n] = val;  // transposed
    }
  }
}

// ---------------- K2: column sums of V per (b,g,dch) ----------------------------
__global__ __launch_bounds__(128) void vcolsum_kernel(
    const unsigned short* __restrict__ vt, float* __restrict__ vcol) {
  int bg = blockIdx.x;
  int t = threadIdx.x;
  if (t >= D_) return;
  const unsigned short* row = vt + ((size_t)bg * D_ + t) * N_;
  float s = 0.f;
  for (int n = 0; n < N_; ++n) s += bf2f(row[n]);
  vcol[bg * D_ + t] = s;
}

// ---------------- K3: QK^T + 2-pass online softmax + cross-head mix -------------
// block = 8 waves; wave w owns head h=w for one 16-query tile; heads mixed in LDS.
__global__ __launch_bounds__(256) void qk_softmax_mix_kernel(
    const unsigned short* __restrict__ qb, const unsigned short* __restrict__ kb,
    const float* __restrict__ rw, const float* __restrict__ bn_gamma,
    unsigned short* __restrict__ ap) {
  __shared__ float ptile[H_][256];
  const float scale = 0.1020620726159658f;  // 96^-0.5
  int tid = threadIdx.x;
  int w = tid >> 5;        // wave id == head id
  int lane = tid & 31;
  int half = lane >> 4;
  int l16 = lane & 15;
  int blk = blockIdx.x;
  int b = blk >> 6;        // 64 query-tiles per batch
  int q0 = (blk & 63) << 4;

  float rwg[H_];           // gamma'[g] * rw[g,h] with g = w (uniform per wave)
  {
    float gp = bn_gamma[w] * rsqrtf(1.0f + BN_EPS);
    for (int h = 0; h < H_; ++h) rwg[h] = rw[w * H_ + h] * gp;
  }

  const unsigned short* qrow =
      qb + ((size_t)(b * H_ + w) * N_ + (q0 + l16)) * D_;
  v16bf aq0 = ldA(qrow, 0, half);
  v16bf aq1 = ldA(qrow, 32, half);
  v16bf aq2 = ldA(qrow, 64, half);
  const unsigned short* kbase = kb + (size_t)(b * H_ + w) * N_ * D_;

  float M[8], L[8];
  for (int r = 0; r < 8; ++r) { M[r] = -3.0e38f; L[r] = 0.f; }

  // pass 1: per-lane online max/sum over this lane's column subset
  for (int kt = 0; kt < 64; ++kt) {
    const unsigned short* krow = kbase + (size_t)(kt * 16 + l16) * D_;
    __builtin_prefetch(krow + 16 * D_, 0, 1);
    v8f acc = {0.f, 0.f, 0.f, 0.f, 0.f, 0.f, 0.f, 0.f};
    acc = wmma_bf16(aq0, ldB(krow, 0, half), acc);
    acc = wmma_bf16(aq1, ldB(krow, 32, half), acc);
    acc = wmma_bf16(aq2, ldB(krow, 64, half), acc);
    for (int r = 0; r < 8; ++r) {
      float s = acc[r] * scale;
      float mn = fmaxf(M[r], s);
      L[r] = L[r] * __expf(M[r] - mn) + __expf(s - mn);
      M[r] = mn;
    }
  }
  // combine the 16 lanes of each half (xor masks 1,2,4,8 stay within a half)
  for (int r = 0; r < 8; ++r) {
    float m = M[r], l = L[r];
    for (int off = 1; off < 16; off <<= 1) {
      float mo = __shfl_xor(m, off, 32);
      float lo = __shfl_xor(l, off, 32);
      float mn = fmaxf(m, mo);
      l = l * __expf(m - mn) + lo * __expf(mo - mn);
      m = mn;
    }
    M[r] = m;
    L[r] = (l > 0.f) ? 1.0f / l : 0.f;  // keep reciprocal
  }

  // pass 2: recompute S, normalize, mix heads through LDS, emit bf16 attn'
  unsigned short* abase = ap + (size_t)(b * H_ + w) * N_ * N_;
  for (int kt = 0; kt < 64; ++kt) {
    const unsigned short* krow = kbase + (size_t)(kt * 16 + l16) * D_;
    v8f acc = {0.f, 0.f, 0.f, 0.f, 0.f, 0.f, 0.f, 0.f};
    acc = wmma_bf16(aq0, ldB(krow, 0, half), acc);
    acc = wmma_bf16(aq1, ldB(krow, 32, half), acc);
    acc = wmma_bf16(aq2, ldB(krow, 64, half), acc);
    for (int r = 0; r < 8; ++r)
      ptile[w][r * 32 + lane] = __expf(acc[r] * scale - M[r]) * L[r];
    __syncthreads();
    for (int r = 0; r < 8; ++r) {
      float mix = 0.f;
      for (int h = 0; h < H_; ++h) mix += rwg[h] * ptile[h][r * 32 + lane];
      int m = r + 8 * half;
      abase[(size_t)(q0 + m) * N_ + kt * 16 + l16] = f2bf(mix);
    }
    __syncthreads();
  }
}

// ---------------- K4: X = attn' @ V + (gamma'*rb+beta)*colsum(V) ----------------
__global__ __launch_bounds__(256) void pv_kernel(
    const unsigned short* __restrict__ ap, const unsigned short* __restrict__ vt,
    const float* __restrict__ vcol, const float* __restrict__ rb,
    const float* __restrict__ bn_gamma, const float* __restrict__ bn_beta,
    unsigned short* __restrict__ xb) {
  int tid = threadIdx.x;
  int w = tid >> 5, lane = tid & 31, half = lane >> 4, l16 = lane & 15;
  int task = blockIdx.x * 8 + w;  // 2048 wave-tasks
  int b = task >> 9;
  int rem = task & 511;
  int g = rem >> 6;
  int q0 = (rem & 63) << 4;

  const unsigned short* arow =
      ap + (size_t)(b * H_ + g) * N_ * N_ + (size_t)(q0 + l16) * N_;
  const unsigned short* vbase = vt + (size_t)(b * H_ + g) * D_ * N_;

  v8f acc[6];
  v8f zero = {0.f, 0.f, 0.f, 0.f, 0.f, 0.f, 0.f, 0.f};
  for (int j = 0; j < 6; ++j) acc[j] = zero;
  for (int c = 0; c < 32; ++c) {
    v16bf afrag = ldA(arow, c * 32, half);
    __builtin_prefetch(arow + (c + 1) * 32, 0, 1);
    for (int j = 0; j < 6; ++j) {
      const unsigned short* vrow = vbase + (size_t)(j * 16 + l16) * N_;
      acc[j] = wmma_bf16(afrag, ldB(vrow, c * 32, half), acc[j]);
    }
  }
  float gp = bn_gamma[g] * rsqrtf(1.0f + BN_EPS);
  float cc = gp * rb[g] + bn_beta[g];
  for (int j = 0; j < 6; ++j) {
    int dch = j * 16 + l16;
    float cterm = cc * vcol[(b * H_ + g) * D_ + dch];
    for (int r = 0; r < 8; ++r) {
      int qrow = q0 + r + 8 * half;
      xb[(size_t)(b * N_ + qrow) * C_ + g * D_ + dch] = f2bf(acc[j][r] + cterm);
    }
  }
}

// ---------------- K5: cast projection weight to bf16 ----------------------------
__global__ void castpw_kernel(const float* __restrict__ pw,
                              unsigned short* __restrict__ pwb) {
  int i = blockIdx.x * 256 + threadIdx.x;
  if (i < C_ * C_) pwb[i] = f2bf(pw[i]);
}

// ---------------- K6: out = X @ pw^T + pb (fp32 out) ----------------------------
__global__ __launch_bounds__(256) void proj_kernel(
    const unsigned short* __restrict__ xb, const unsigned short* __restrict__ pwb,
    const float* __restrict__ pb, float* __restrict__ out) {
  int tid = threadIdx.x;
  int w = tid >> 5, lane = tid & 31, half = lane >> 4, l16 = lane & 15;
  int task = blockIdx.x * 8 + w;  // 1536 wave-tasks: 256 row tiles x 6 col groups
  int rowtile = task / 6;
  int cg = task % 6;
  int row0 = rowtile * 16;
  const unsigned short* arow = xb + (size_t)(row0 + l16) * C_;
  v8f acc[8];
  v8f zero = {0.f, 0.f, 0.f, 0.f, 0.f, 0.f, 0.f, 0.f};
  for (int j = 0; j < 8; ++j) acc[j] = zero;
  for (int c = 0; c < 24; ++c) {
    v16bf afrag = ldA(arow, c * 32, half);
    __builtin_prefetch(arow + (c + 1) * 32, 0, 1);
    for (int j = 0; j < 8; ++j) {
      const unsigned short* brow = pwb + (size_t)(cg * 128 + j * 16 + l16) * C_;
      acc[j] = wmma_bf16(afrag, ldB(brow, c * 32, half), acc[j]);
    }
  }
  for (int j = 0; j < 8; ++j) {
    int co = cg * 128 + j * 16 + l16;
    float bias = pb[co];
    for (int r = 0; r < 8; ++r) {
      int row = row0 + r + 8 * half;
      out[(size_t)row * C_ + co] = acc[j][r] + bias;
    }
  }
}

extern "C" void kernel_launch(void* const* d_in, const int* in_sizes, int n_in,
                              void* d_out, int out_size, void* d_ws, size_t ws_size,
                              hipStream_t stream) {
  (void)in_sizes; (void)n_in; (void)out_size; (void)ws_size;
  const float* q  = (const float*)d_in[0];
  const float* k  = (const float*)d_in[1];
  const float* v  = (const float*)d_in[2];
  const float* qw = (const float*)d_in[3];
  const float* kw = (const float*)d_in[4];
  const float* vw = (const float*)d_in[5];
  const float* rw = (const float*)d_in[6];
  const float* rb = (const float*)d_in[7];
  const float* gg = (const float*)d_in[8];
  const float* bb = (const float*)d_in[9];
  const float* pw = (const float*)d_in[10];
  const float* pb = (const float*)d_in[11];
  float* out = (float*)d_out;

  // workspace layout (region reuse keeps total at ~86 MB):
  //   [0,6M)      qb   (K1->K3)    then xb  (K4->K6)
  //   [6M,12M)    kb   (K1->K3)    then pwb (K5->K6)
  //   [12M,18M)   vt   (K1->K2,K4)
  //   [18M,82M+)  ap   (K3->K4)    64 MB bf16 attn'
  //   [~86M)      vcol (K2->K4)
  char* ws = (char*)d_ws;
  unsigned short* qb  = (unsigned short*)(ws + 0);
  unsigned short* kb  = (unsigned short*)(ws + 6291456);
  unsigned short* vt  = (unsigned short*)(ws + 12582912);
  unsigned short* ap  = (unsigned short*)(ws + 18874368);
  unsigned short* xb  = (unsigned short*)(ws + 0);         // reuse qb
  unsigned short* pwb = (unsigned short*)(ws + 6291456);   // reuse kb (after K3)
  float*          vcol = (float*)(ws + 85983232);

  conv_cast_kernel<<<B_ * N_, 256, 0, stream>>>(q, k, v, qw, kw, vw, qb, kb, vt);
  vcolsum_kernel<<<B_ * H_, 128, 0, stream>>>(vt, vcol);
  qk_softmax_mix_kernel<<<B_ * (N_ / 16), 256, 0, stream>>>(qb, kb, rw, gg, ap);
  castpw_kernel<<<(C_ * C_ + 255) / 256, 256, 0, stream>>>(pw, pwb);
  pv_kernel<<<256, 256, 0, stream>>>(ap, vt, vcol, rb, gg, bb, xb);
  proj_kernel<<<192, 256, 0, stream>>>(xb, pwb, pb, out);
}
